// EncoderFrame_5875515261250
// MI455X (gfx1250) — matrix operand order repaired and checked by reading, not compiled, not run on hardware
//
#include <hip/hip_runtime.h>
#include <hip/hip_bf16.h>

// Problem dims
#define B_     256
#define S_     64
#define D_INP  1024       // D_SLOT + D_TEXT
#define DSLOT  256
#define DTEXT  768
#define H_     512
#define H3     1536
#define E_     512
#define ROWS   (S_ * B_)  // 16384

typedef __attribute__((ext_vector_type(4)))  unsigned int u32x4;
typedef __attribute__((ext_vector_type(16))) __bf16       v16bf;
typedef __attribute__((ext_vector_type(8)))  float        v8f;
typedef __attribute__((ext_vector_type(4)))  unsigned int tdm_u4;
typedef __attribute__((ext_vector_type(8)))  int          tdm_i8;
typedef __attribute__((ext_vector_type(4)))  int          tdm_i4;

union FragBF { u32x4 u[2]; v16bf v; };

#define TILE_ELEMS (64 * 32)   // one staging tile: 64 rows x 32 bf16

__device__ __forceinline__ unsigned short f2bf(float f) {
  unsigned int u = __float_as_uint(f);
  u += 0x7fffu + ((u >> 16) & 1u);          // round-to-nearest-even
  return (unsigned short)(u >> 16);
}

// Read 16B from LDS via an explicit addrspace(3) pointer built from the
// 32-bit LDS offset (low 32 bits of the generic address).  Guarantees
// ds_load_b128 lowering and hides provenance (LDS is written by the TDM,
// which the compiler cannot see).
__device__ __forceinline__ u32x4 lds_read16(const unsigned short* p) {
  return *(const __attribute__((address_space(3))) u32x4*)(unsigned)(size_t)p;
}

// ---------------------------------------------------------------------------
// TDM: DMA 64 rows x 32 bf16 (64B/row) of a row-major [*, stride] matrix into
// LDS as one contiguous 4KB tile.  2D descriptor: groups 2/3 zero.
// Tracked by TENSORcnt.  All inputs are wave-uniform scalars (no threadIdx),
// so descriptor groups are built in SALU.
// ---------------------------------------------------------------------------
__device__ __forceinline__ void tdm_load_tile64(unsigned lds_off_bytes,
                                                const unsigned short* g,
                                                unsigned stride_elems) {
  unsigned long long ga = (unsigned long long)(size_t)g;
  tdm_u4 g0;
  g0[0] = 1u;                                              // count=1, user mode
  g0[1] = lds_off_bytes;                                   // lds_addr
  g0[2] = (unsigned)ga;                                    // global_addr[31:0]
  g0[3] = (unsigned)((ga >> 32) & 0x01ffffffu) | (2u << 30); // addr[56:32] | type=2
  tdm_i8 g1;
  const unsigned td0 = stride_elems;                       // tensor_dim0
  const unsigned td1 = 1u << 20;                           // tensor_dim1 (no OOB)
  g1[0] = (int)(1u << 16);                                 // data_size = 2 bytes
  g1[1] = (int)((td0 & 0xffffu) << 16);                    // tensor_dim0 lo
  g1[2] = (int)(((td0 >> 16) & 0xffffu) | ((td1 & 0xffffu) << 16));
  g1[3] = (int)(((td1 >> 16) & 0xffffu) | (32u << 16));    // tile_dim0 = 32
  g1[4] = (int)64u;                                        // tile_dim1 = 64, tile_dim2 = 0
  g1[5] = (int)stride_elems;                               // tensor_dim0_stride[31:0]
  g1[6] = 0;
  g1[7] = 0;
  tdm_i4 z4 = {0, 0, 0, 0};
  tdm_i8 z8 = {0, 0, 0, 0, 0, 0, 0, 0};
  __builtin_amdgcn_tensor_load_to_lds(g0, g1, z4, z4, z8, 0);
}

// Stage one 64x32 A tile + one 64x32 B tile into LDS buffer `buf`.
// Caller guards so only wave 0 issues (branch, not exec-predication risk:
// waves with EXEC==0 take s_cbranch_execz around the tensor op entirely).
__device__ __forceinline__ void stage_tiles(unsigned short* As, unsigned short* Bs,
                                            int buf,
                                            const unsigned short* A, int lda, int m0,
                                            const unsigned short* Bm, int ldb, int n0,
                                            int kc) {
  unsigned aoff = (unsigned)(size_t)(As + buf * TILE_ELEMS);
  unsigned boff = (unsigned)(size_t)(Bs + buf * TILE_ELEMS);
  tdm_load_tile64(aoff, A + (size_t)m0 * lda + kc, (unsigned)lda);
  tdm_load_tile64(boff, Bm + (size_t)n0 * ldb + kc, (unsigned)ldb);
}

// ---------------------------------------------------------------------------
// Block-tiled bf16 GEMM accumulate, C block tile 64(M) x 64(N), 8 waves.
// Wave tile 16x32: one A frag reused against two B frags -> 2 WMMA / k-chunk.
// Double-buffered LDS staging via TDM (wave 0 issues) so the DMA of chunk k+1
// overlaps the ds_load + v_wmma stream of chunk k.
// Bm is row-major [N][K] (weights as given, computing x @ W^T).
// ---------------------------------------------------------------------------
__device__ __forceinline__ void gemm_accum(
    const unsigned short* __restrict__ A, int lda, int m0,
    const unsigned short* __restrict__ Bm, int ldb, int n0,
    int K, unsigned short* As, unsigned short* Bs, v8f& c0, v8f& c1)
{
  const int tid  = threadIdx.x;
  const int lane = tid & 31;
  const int wave = tid >> 5;
  const int mw = (wave >> 1) << 4;          // 0,16,32,48
  const int nw = (wave & 1) << 5;           // 0,32
  const int lr = lane & 15;
  const int ka = (lane < 16) ? 0 : 16;      // 16-bit A frag K-half per lane group
  const int kb = (lane < 16) ? 0 : 16;      // 16-bit B frag K-half per lane group

  if (wave == 0)
    stage_tiles(As, Bs, 0, A, lda, m0, Bm, ldb, n0, 0);
  __builtin_amdgcn_s_wait_tensorcnt(0);     // no-op for waves with TENSORcnt==0
  __syncthreads();

  int cur = 0;
  for (int kc = 0; kc < K; kc += 32) {
    if (wave == 0 && kc + 32 < K)  // prefetch next chunk into idle buffer (DMA)
      stage_tiles(As, Bs, cur ^ 1, A, lda, m0, Bm, ldb, n0, kc + 32);

    const unsigned short* as = As + cur * TILE_ELEMS;
    const unsigned short* bs = Bs + cur * TILE_ELEMS;
    FragBF a, b0, b1;
    // A 16x32 frag: lane<16 holds M=lr, K=0..15; lane>=16 K=16..31
    a.u[0]  = lds_read16(as + (mw + lr) * 32 + ka);
    a.u[1]  = lds_read16(as + (mw + lr) * 32 + ka + 8);
    // B 32x16 frags: lane<16 holds N=lr, K=0..15; lane>=16 K=16..31
    b0.u[0] = lds_read16(bs + (nw + lr) * 32 + kb);
    b0.u[1] = lds_read16(bs + (nw + lr) * 32 + kb + 8);
    b1.u[0] = lds_read16(bs + (nw + 16 + lr) * 32 + kb);
    b1.u[1] = lds_read16(bs + (nw + 16 + lr) * 32 + kb + 8);

    c0 = __builtin_amdgcn_wmma_f32_16x16x32_bf16(false, a.v, false, b0.v,
                                                 (short)0, c0, false, false);
    c1 = __builtin_amdgcn_wmma_f32_16x16x32_bf16(false, a.v, false, b1.v,
                                                 (short)0, c1, false, false);

    __builtin_amdgcn_s_wait_tensorcnt(0);   // DMA of next chunk done (wave 0)
    __syncthreads();                        // visible to all waves; cur reads done
    cur ^= 1;
  }
}

// ---------------------------------------------------------------------------
// fp32 -> bf16 weight conversion
// ---------------------------------------------------------------------------
__global__ void k_cvt_bf16(const float* __restrict__ src,
                           unsigned short* __restrict__ dst, int n) {
  int base = blockIdx.x * 1024 + threadIdx.x;
#pragma unroll
  for (int j = 0; j < 4; ++j) {
    int idx = base + j * 256;
    if (idx < n) dst[idx] = f2bf(src[idx]);
  }
}

// ---------------------------------------------------------------------------
// Build X_bf16 [S][B][D_INP]: concat(embed_table[slot], value_embed), bf16
// ---------------------------------------------------------------------------
__global__ void k_build_x(const int* __restrict__ slot_ids,
                          const float* __restrict__ value_embeds,
                          const float* __restrict__ embed_table,
                          unsigned short* __restrict__ Xbf) {
  const int bs = blockIdx.x;            // b*S + t  (input layout [B,S])
  const int b  = bs / S_;
  const int t  = bs % S_;
  const int slot = slot_ids[bs];
  const float* emb = embed_table + (size_t)slot * DSLOT;
  const float* val = value_embeds + (size_t)bs * DTEXT;
  unsigned short* dst = Xbf + ((size_t)t * B_ + b) * D_INP;  // time-major rows
  for (int j = threadIdx.x; j < D_INP; j += 256) {
    float f = (j < DSLOT) ? emb[j] : val[j - DSLOT];
    dst[j] = f2bf(f);
  }
}

__global__ void k_init_h(float* __restrict__ hf, unsigned short* __restrict__ hb) {
  int i = blockIdx.x * 256 + threadIdx.x;
  if (i < 2 * B_ * H_) { hf[i] = 0.0f; hb[i] = 0; }
}

// ---------------------------------------------------------------------------
// One GRU step, both directions (blockIdx.z = dir):
//   Gx = x_tx @ W_ih^T + b_ih      (K = 1024)
//   Gh = h    @ W_hh^T + b_hh      (K = 512)
// kept separate because n-gate needs r * (h-path).
// ---------------------------------------------------------------------------
__global__ void k_step_gemm(const unsigned short* __restrict__ Xbf,
                            const unsigned short* __restrict__ Wih,  // [2][H3][D_INP]
                            const unsigned short* __restrict__ Whh,  // [2][H3][H_]
                            const unsigned short* __restrict__ hbf,  // [2][B_][H_]
                            const float* __restrict__ b_ih_f,
                            const float* __restrict__ b_hh_f,
                            const float* __restrict__ b_ih_b,
                            const float* __restrict__ b_hh_b,
                            float* __restrict__ Gx, float* __restrict__ Gh,
                            int t) {
  __shared__ unsigned short As[2 * TILE_ELEMS];
  __shared__ unsigned short Bs[2 * TILE_ELEMS];

  const int dir = blockIdx.z;
  const int tx  = dir ? (S_ - 1 - t) : t;
  const int m0  = blockIdx.x * 64;
  const int n0  = blockIdx.y * 64;

  const unsigned short* A1 = Xbf + (size_t)tx * B_ * D_INP;
  const unsigned short* B1 = Wih + (size_t)dir * H3 * D_INP;
  const unsigned short* A2 = hbf + (size_t)dir * B_ * H_;
  const unsigned short* B2 = Whh + (size_t)dir * H3 * H_;

  v8f cx0 = {0.f,0.f,0.f,0.f,0.f,0.f,0.f,0.f};
  v8f cx1 = {0.f,0.f,0.f,0.f,0.f,0.f,0.f,0.f};
  v8f ch0 = {0.f,0.f,0.f,0.f,0.f,0.f,0.f,0.f};
  v8f ch1 = {0.f,0.f,0.f,0.f,0.f,0.f,0.f,0.f};
  gemm_accum(A1, D_INP, m0, B1, D_INP, n0, D_INP, As, Bs, cx0, cx1);
  gemm_accum(A2, H_,    m0, B2, H_,    n0, H_,    As, Bs, ch0, ch1);

  const float* bih = dir ? b_ih_b : b_ih_f;
  const float* bhh = dir ? b_hh_b : b_hh_f;

  const int lane = threadIdx.x & 31;
  const int wave = threadIdx.x >> 5;
  const int mw = (wave >> 1) << 4;
  const int nw = (wave & 1) << 5;
  const int na = n0 + nw + (lane & 15);
  const int nb = na + 16;
  const float bi0 = bih[na], bh0 = bhh[na];
  const float bi1 = bih[nb], bh1 = bhh[nb];
  const int mrow = m0 + mw + ((lane >> 4) << 3);   // C layout: VGPRv -> M = v (+8 hi half)
  float* gx = Gx + (size_t)dir * B_ * H3;
  float* gh = Gh + (size_t)dir * B_ * H3;
#pragma unroll
  for (int v = 0; v < 8; ++v) {
    const int m = mrow + v;
    gx[(size_t)m * H3 + na] = cx0[v] + bi0;
    gx[(size_t)m * H3 + nb] = cx1[v] + bi1;
    gh[(size_t)m * H3 + na] = ch0[v] + bh0;
    gh[(size_t)m * H3 + nb] = ch1[v] + bh1;
  }
}

// ---------------------------------------------------------------------------
// Elementwise GRU gates: r,z = sigmoid; n = tanh(gx_n + r*gh_n);
// h' = (1-z)*n + z*h.  Writes f32 state, bf16 state, and bf16 output row.
// ---------------------------------------------------------------------------
__global__ void k_gates(const float* __restrict__ Gx, const float* __restrict__ Gh,
                        float* __restrict__ hf, unsigned short* __restrict__ hbf,
                        unsigned short* __restrict__ Hcat, int t) {
  const int idx = blockIdx.x * 256 + threadIdx.x;
  if (idx >= 2 * B_ * H_) return;
  const int dir = idx >> 17;                 // / (256*512)
  const int rem = idx & (B_ * H_ - 1);
  const int b   = rem >> 9;
  const int j   = rem & (H_ - 1);
  const size_t row = ((size_t)dir * B_ + b) * H3;

  const float sr  = Gx[row + j]          + Gh[row + j];
  const float sz  = Gx[row + H_ + j]     + Gh[row + H_ + j];
  const float gxn = Gx[row + 2 * H_ + j];
  const float ghn = Gh[row + 2 * H_ + j];
  const float h   = hf[idx];

  const float r  = 1.0f / (1.0f + __expf(-sr));
  const float z  = 1.0f / (1.0f + __expf(-sz));
  const float nn = tanhf(gxn + r * ghn);
  const float hnew = (1.0f - z) * nn + z * h;

  hf[idx] = hnew;
  const unsigned short hb = f2bf(hnew);
  hbf[idx] = hb;
  const int tx = dir ? (S_ - 1 - t) : t;
  Hcat[((size_t)tx * B_ + b) * (2 * H_) + (size_t)dir * H_ + j] = hb;
}

// ---------------------------------------------------------------------------
// Final projection: [ROWS, 2H] @ W_proj^T -> d_out [B][S][E] (f32)
// ---------------------------------------------------------------------------
__global__ void k_proj(const unsigned short* __restrict__ Hcat,
                       const unsigned short* __restrict__ Wp,   // [E][2H] bf16
                       float* __restrict__ out) {
  __shared__ unsigned short As[2 * TILE_ELEMS];
  __shared__ unsigned short Bs[2 * TILE_ELEMS];
  const int m0 = blockIdx.x * 64;
  const int n0 = blockIdx.y * 64;

  v8f c0 = {0.f,0.f,0.f,0.f,0.f,0.f,0.f,0.f};
  v8f c1 = {0.f,0.f,0.f,0.f,0.f,0.f,0.f,0.f};
  gemm_accum(Hcat, 2 * H_, m0, Wp, 2 * H_, n0, 2 * H_, As, Bs, c0, c1);

  const int lane = threadIdx.x & 31;
  const int wave = threadIdx.x >> 5;
  const int mw = (wave >> 1) << 4;
  const int nw = (wave & 1) << 5;
  const int ea = n0 + nw + (lane & 15);
  const int eb = ea + 16;
  const int mrow = m0 + mw + ((lane >> 4) << 3);
#pragma unroll
  for (int v = 0; v < 8; ++v) {
    const int m = mrow + v;        // m = s*B + b (time-major rows)
    const int s = m >> 8;
    const int b = m & (B_ - 1);
    out[((size_t)b * S_ + s) * E_ + ea] = c0[v];
    out[((size_t)b * S_ + s) * E_ + eb] = c1[v];
  }
}

// ---------------------------------------------------------------------------
extern "C" void kernel_launch(void* const* d_in, const int* in_sizes, int n_in,
                              void* d_out, int out_size, void* d_ws, size_t ws_size,
                              hipStream_t stream) {
  (void)in_sizes; (void)n_in; (void)out_size; (void)ws_size;

  const int*   slot_ids     = (const int*)d_in[0];
  const float* value_embeds = (const float*)d_in[1];
  const float* embed_table  = (const float*)d_in[2];
  const float* W_ih_f = (const float*)d_in[3];
  const float* W_hh_f = (const float*)d_in[4];
  const float* b_ih_f = (const float*)d_in[5];
  const float* b_hh_f = (const float*)d_in[6];
  const float* W_ih_b = (const float*)d_in[7];
  const float* W_hh_b = (const float*)d_in[8];
  const float* b_ih_b = (const float*)d_in[9];
  const float* b_hh_b = (const float*)d_in[10];
  const float* W_proj = (const float*)d_in[11];
  float* out = (float*)d_out;

  char* ws = (char*)d_ws;
  size_t off = 0;
  auto alloc = [&](size_t bytes) -> void* {
    void* p = ws + off;
    off = (off + bytes + 255) & ~(size_t)255;
    return p;
  };
  unsigned short* Xbf   = (unsigned short*)alloc((size_t)ROWS * D_INP * 2);
  unsigned short* WihBf = (unsigned short*)alloc((size_t)2 * H3 * D_INP * 2);
  unsigned short* WhhBf = (unsigned short*)alloc((size_t)2 * H3 * H_ * 2);
  unsigned short* WpBf  = (unsigned short*)alloc((size_t)E_ * 2 * H_ * 2);
  unsigned short* Hcat  = (unsigned short*)alloc((size_t)ROWS * 2 * H_ * 2);
  float*          Gx    = (float*)alloc((size_t)2 * B_ * H3 * 4);
  float*          Gh    = (float*)alloc((size_t)2 * B_ * H3 * 4);
  float*          hf    = (float*)alloc((size_t)2 * B_ * H_ * 4);
  unsigned short* hbf   = (unsigned short*)alloc((size_t)2 * B_ * H_ * 2);

  auto cvt = [&](const float* s, unsigned short* d, int n) {
    k_cvt_bf16<<<(n + 1023) / 1024, 256, 0, stream>>>(s, d, n);
  };
  cvt(W_ih_f, WihBf,                       H3 * D_INP);
  cvt(W_ih_b, WihBf + (size_t)H3 * D_INP,  H3 * D_INP);
  cvt(W_hh_f, WhhBf,                       H3 * H_);
  cvt(W_hh_b, WhhBf + (size_t)H3 * H_,     H3 * H_);
  cvt(W_proj, WpBf,                        E_ * 2 * H_);

  k_build_x<<<ROWS, 256, 0, stream>>>(slot_ids, value_embeds, embed_table, Xbf);
  k_init_h<<<(2 * B_ * H_ + 255) / 256, 256, 0, stream>>>(hf, hbf);

  for (int t = 0; t < S_; ++t) {
    k_step_gemm<<<dim3(B_ / 64, H3 / 64, 2), 256, 0, stream>>>(
        Xbf, WihBf, WhhBf, hbf, b_ih_f, b_hh_f, b_ih_b, b_hh_b, Gx, Gh, t);
    k_gates<<<(2 * B_ * H_) / 256, 256, 0, stream>>>(Gx, Gh, hf, hbf, Hcat, t);
  }

  k_proj<<<dim3(ROWS / 64, E_ / 64), 256, 0, stream>>>(Hcat, WpBf, out);
}
